// RNNfromscratch_75926431859009
// MI455X (gfx1250) — compile-verified, matrix operand order checked by reading, not run on hardware
//
#include <hip/hip_runtime.h>
#include <hip/hip_bf16.h>

// ---------------------------------------------------------------------------
// Problem constants (from reference): S=512, B=64, I=256, H=1024, L=3
// ---------------------------------------------------------------------------
#define RS 512
#define RB 64
#define RI 256
#define RH 1024
#define RL 3

#define NTILE 64                       // output columns per workgroup
#define NBLOCKS (RL * (RH / NTILE))    // 3 * 16 = 48 persistent workgroups
#define TPB 128                        // 4 wave32s per workgroup
#define KC 256                         // K-chunk staged in LDS (32 KB / chunk)

typedef __bf16 bf16_t;
typedef __attribute__((ext_vector_type(16))) __bf16 v16bf;
typedef __attribute__((ext_vector_type(8)))  __bf16 v8bf;
typedef __attribute__((ext_vector_type(8)))  float  v8f;

// ---- workspace layout (bytes) ---------------------------------------------
#define OFF_CTR   0u
#define OFF_HBUF  256u                                    // 2 * L*B*H bf16
#define HBUF_BYTES (2u * RL * RB * RH * 2u)               // 786432
#define OFF_XBF   (OFF_HBUF + HBUF_BYTES)                 // S*B*I bf16
#define XBF_BYTES (RS * RB * RI * 2u)                     // 16 MB
#define OFF_WXHT  (OFF_XBF + XBF_BYTES)                   // [H][I] bf16
#define WXHT_BYTES (RH * RI * 2u)
#define OFF_WHHT  (OFF_WXHT + WXHT_BYTES)                 // [H][H] bf16
#define MAT_BYTES (RH * RH * 2u)
#define OFF_WT    (OFF_WHHT + MAT_BYTES)                  // 2 x [H][H]
#define OFF_UT    (OFF_WT + 2u * MAT_BYTES)               // 2 x [H][H]

// ---------------------------------------------------------------------------
// helpers
// ---------------------------------------------------------------------------
__device__ __forceinline__ bf16_t f2bf(float f) {
  unsigned u = __float_as_uint(f);
  unsigned r = (u + 0x7FFFu + ((u >> 16) & 1u)) >> 16;   // round-nearest-even
  unsigned short s = (unsigned short)r;
  return *reinterpret_cast<bf16_t*>(&s);
}

__device__ __forceinline__ float fast_tanh(float x) {
  float e = __expf(2.0f * x);
  return 1.0f - 2.0f / (e + 1.0f);
}

// One 16-bit WMMA fragment per the documented lane layout: this lane supplies
// K = [kg..kg+7] and [kg+16..kg+23] of a 32-wide K chunk, kg = lane<16 ? 0 : 8.
// Works for both global (global_load_b128) and LDS (ds_load_b128) pointers.
__device__ __forceinline__ v16bf load_frag(const bf16_t* p, int kg) {
  v8bf lo = *reinterpret_cast<const v8bf*>(p + kg);
  v8bf hi = *reinterpret_cast<const v8bf*>(p + kg + 16);
  return __builtin_shufflevector(lo, hi, 0, 1, 2, 3, 4, 5, 6, 7,
                                         8, 9, 10, 11, 12, 13, 14, 15);
}

// Async-copy one [64][KC] bf16 activation chunk (rows = batch, global row
// stride Kdim) into LDS, laid out [64][KC]. 2048 16-byte units over 128
// threads -> 16 async b128 loads per thread (ASYNCcnt += 16 per wave).
__device__ __forceinline__ void async_stage_chunk(const bf16_t* __restrict__ act,
                                                  int Kdim, int kbase,
                                                  bf16_t* shdst, int tid) {
#pragma unroll
  for (int i = 0; i < 16; ++i) {
    int fi  = i * TPB + tid;       // 0..2047
    int row = fi >> 5;             // 32 units of 16B per row
    int ku  = (fi & 31) * 8;       // bf16 element offset within row
    const bf16_t* g = act + (size_t)row * Kdim + kbase + ku;
    unsigned lds = (unsigned)(size_t)(shdst + row * KC + ku);  // LDS = addr[31:0]
    asm volatile("global_load_async_to_lds_b128 %0, %1, off"
                 :: "v"(lds), "v"(g) : "memory");
  }
}

// ---------------------------------------------------------------------------
// prep kernels
// ---------------------------------------------------------------------------
__global__ void init_ws_kernel(int* __restrict__ ctr, unsigned* __restrict__ hbuf,
                               int hwords) {
  int i = blockIdx.x * blockDim.x + threadIdx.x;
  if (i == 0) *ctr = 0;
  for (int j = i; j < hwords; j += gridDim.x * blockDim.x) hbuf[j] = 0u;
}

__global__ void cvt_bf16_kernel(const float* __restrict__ in,
                                bf16_t* __restrict__ out, int n) {
  int i = blockIdx.x * blockDim.x + threadIdx.x;
  for (int j = i; j < n; j += gridDim.x * blockDim.x) out[j] = f2bf(in[j]);
}

// in: [K][N] f32, out: [N][K] bf16  (coalesced writes)
__global__ void transpose_bf16_kernel(const float* __restrict__ in,
                                      bf16_t* __restrict__ out, int K, int N) {
  int i = blockIdx.x * blockDim.x + threadIdx.x;
  int total = K * N;
  for (int o = i; o < total; o += gridDim.x * blockDim.x) {
    int n = o / K;
    int k = o - n * K;
    out[o] = f2bf(in[(size_t)k * N + n]);
  }
}

// ---------------------------------------------------------------------------
// persistent RNN kernel: double-buffered LDS A-staging via async global->LDS,
// WMMA compute, one grid barrier per timestep, double-buffered h
// ---------------------------------------------------------------------------
__global__ __launch_bounds__(TPB) void rnn_persistent_kernel(
    const bf16_t* __restrict__ xbf,     // [S][B][I]
    const bf16_t* __restrict__ wxhT,    // [H][I]
    const bf16_t* __restrict__ whhT,    // [H][H]
    const bf16_t* __restrict__ wT,      // [2][H][H]  (W_layers^T)
    const bf16_t* __restrict__ uT,      // [2][H][H]  (U_layers^T)
    const float* __restrict__ b_h,      // [H]
    const float* __restrict__ b_layers, // [2][H]
    bf16_t* __restrict__ hbuf,          // 2 x [L][B][H]
    int* __restrict__ ctr,
    float* __restrict__ out)            // [S][B][H] ++ [L][B][H]
{
  __shared__ bf16_t sh[2 * RB * KC];    // 2 x 32 KB double buffer

  const int layer  = blockIdx.x / (RH / NTILE);
  const int nbase  = (blockIdx.x % (RH / NTILE)) * NTILE;
  const int wave   = threadIdx.x >> 5;
  const int lane   = threadIdx.x & 31;
  const int laneRow = lane & 15;               // row/col within a 16-tile
  const int kg      = (lane < 16) ? 0 : 8;     // K sub-offset for this lane
  const int col     = nbase + wave * 16 + laneRow;   // absolute output column
  const int rbase   = (lane < 16) ? 0 : 8;     // accumulator row base
  const int tid     = threadIdx.x;

  const float bias = (layer == 0) ? b_h[col] : b_layers[(layer - 1) * RH + col];
  const size_t LBH = (size_t)RL * RB * RH;

  // per-layer GEMM segments (activation base is step-dependent only via `cur`)
  const bf16_t* wSeg0;  const bf16_t* wSeg1;
  int K0, K1, actOff0, actOff1;   // activation offsets into `cur` (or X marker)
  if (layer == 0) {
    wSeg0 = wxhT; K0 = RI; actOff0 = -1;            // seg0 = X_t
    wSeg1 = whhT; K1 = RH; actOff1 = 0;             // seg1 = h[0]
  } else {
    wSeg0 = uT + (size_t)(layer - 1) * RH * RH; K0 = RH; actOff0 = layer * RB * RH;
    wSeg1 = wT + (size_t)(layer - 1) * RH * RH; K1 = RH; actOff1 = (layer - 1) * RB * RH;
  }
  const int nch0 = K0 / KC;
  const int nch  = (K0 + K1) / KC;

  for (int t = 0; t < RS; ++t) {
    const bf16_t* cur  = hbuf + (size_t)(t & 1) * LBH;
    bf16_t*       next = hbuf + (size_t)((t + 1) & 1) * LBH;
    const bf16_t* xseg = xbf + (size_t)t * RB * RI;

    v8f acc[4];
#pragma unroll
    for (int m = 0; m < 4; ++m) acc[m] = (v8f){0.f,0.f,0.f,0.f,0.f,0.f,0.f,0.f};

    // prologue: stage chunk 0
    {
      const bf16_t* a0 = (actOff0 < 0) ? xseg : cur + actOff0;
      async_stage_chunk(a0, K0, 0, sh, tid);
    }

    for (int c = 0; c < nch; ++c) {
      // issue chunk c+1 into the other buffer (overlaps compute of chunk c)
      if (c + 1 < nch) {
        int cn = c + 1;
        const bf16_t* a; int Kd, kb;
        if (cn < nch0) { a = (actOff0 < 0) ? xseg : cur + actOff0; Kd = K0; kb = cn * KC; }
        else           { a = cur + actOff1;                        Kd = K1; kb = (cn - nch0) * KC; }
        async_stage_chunk(a, Kd, kb, sh + (cn & 1) * (RB * KC), tid);
        asm volatile("s_wait_asynccnt 0x10" ::: "memory");  // chunk c done
      } else {
        asm volatile("s_wait_asynccnt 0x0" ::: "memory");   // all done
      }
      __syncthreads();   // chunk c visible to all waves

      // weight column slice for chunk c (direct global, unique per wave)
      const bf16_t* wcol;
      if (c < nch0) wcol = wSeg0 + (size_t)col * K0 + c * KC;
      else          wcol = wSeg1 + (size_t)col * K1 + (c - nch0) * KC;

      const bf16_t* shb = sh + (c & 1) * (RB * KC);
#pragma unroll
      for (int k0 = 0; k0 < KC; k0 += 32) {
        v16bf bfrag = load_frag(wcol + k0, kg);              // global_load_b128 x2
#pragma unroll
        for (int m = 0; m < 4; ++m) {
          const bf16_t* arow = shb + (m * 16 + laneRow) * KC + k0;
          v16bf afrag = load_frag(arow, kg);                 // ds_load_b128 x2
          acc[m] = __builtin_amdgcn_wmma_f32_16x16x32_bf16(
              false, afrag, false, bfrag, (short)0, acc[m], false, false);
        }
      }
      __syncthreads();   // buffer (c&1) free for reuse at iteration c+1
    }

    // epilogue: bias + tanh, write bf16 h_next (+ f32 outputs)
#pragma unroll
    for (int m = 0; m < 4; ++m) {
#pragma unroll
      for (int r = 0; r < 8; ++r) {
        int row = m * 16 + rbase + r;                 // batch index 0..63
        float v = fast_tanh(acc[m][r] + bias);
        next[(size_t)layer * RB * RH + (size_t)row * RH + col] = f2bf(v);
        if (layer == RL - 1)
          out[(size_t)t * RB * RH + (size_t)row * RH + col] = v;
        if (t == RS - 1)
          out[(size_t)RS * RB * RH + (size_t)layer * RB * RH +
              (size_t)row * RH + col] = v;
      }
    }

    // ---- grid-wide barrier (one per step; reads+writes both done above) ----
    __threadfence();
    __syncthreads();
    if (threadIdx.x == 0) {
      __hip_atomic_fetch_add(ctr, 1, __ATOMIC_RELEASE, __HIP_MEMORY_SCOPE_AGENT);
      const int target = NBLOCKS * (t + 1);
      while (__hip_atomic_load(ctr, __ATOMIC_RELAXED, __HIP_MEMORY_SCOPE_AGENT)
             < target) {
        __builtin_amdgcn_s_sleep(2);
      }
    }
    __syncthreads();
    __threadfence();
  }
}

// ---------------------------------------------------------------------------
// launcher
// ---------------------------------------------------------------------------
extern "C" void kernel_launch(void* const* d_in, const int* in_sizes, int n_in,
                              void* d_out, int out_size, void* d_ws, size_t ws_size,
                              hipStream_t stream) {
  const float* X    = (const float*)d_in[0];  // [S,B,I]
  const float* Wxh  = (const float*)d_in[1];  // [I,H]
  const float* Whh  = (const float*)d_in[2];  // [H,H]
  const float* bh   = (const float*)d_in[3];  // [H]
  const float* Wl   = (const float*)d_in[4];  // [2,H,H]
  const float* Ul   = (const float*)d_in[5];  // [2,H,H]
  const float* bl   = (const float*)d_in[6];  // [2,H]

  char* ws = (char*)d_ws;
  int*    ctr  = (int*)(ws + OFF_CTR);
  bf16_t* hbuf = (bf16_t*)(ws + OFF_HBUF);
  bf16_t* xbf  = (bf16_t*)(ws + OFF_XBF);
  bf16_t* wxhT = (bf16_t*)(ws + OFF_WXHT);
  bf16_t* whhT = (bf16_t*)(ws + OFF_WHHT);
  bf16_t* wT   = (bf16_t*)(ws + OFF_WT);
  bf16_t* uT   = (bf16_t*)(ws + OFF_UT);

  // 1) init barrier counter + zero both h buffers (h0 = zeros)
  {
    int hwords = (int)(HBUF_BYTES / 4);
    init_ws_kernel<<<256, 256, 0, stream>>>(ctr, (unsigned*)hbuf, hwords);
  }
  // 2) X -> bf16
  {
    int n = RS * RB * RI;
    cvt_bf16_kernel<<<1024, 256, 0, stream>>>(X, xbf, n);
  }
  // 3) weight transposes -> bf16 [N][K]
  transpose_bf16_kernel<<<512, 256, 0, stream>>>(Wxh, wxhT, RI, RH);
  transpose_bf16_kernel<<<1024, 256, 0, stream>>>(Whh, whhT, RH, RH);
  transpose_bf16_kernel<<<1024, 256, 0, stream>>>(Wl,            wT,            RH, RH);
  transpose_bf16_kernel<<<1024, 256, 0, stream>>>(Wl + RH * RH,  wT + RH * RH,  RH, RH);
  transpose_bf16_kernel<<<1024, 256, 0, stream>>>(Ul,            uT,            RH, RH);
  transpose_bf16_kernel<<<1024, 256, 0, stream>>>(Ul + RH * RH,  uT + RH * RH,  RH, RH);

  // 4) persistent recurrence
  rnn_persistent_kernel<<<NBLOCKS, TPB, 0, stream>>>(
      xbf, wxhT, whhT, wT, uT, bh, bl, hbuf, ctr, (float*)d_out);
}
